// SRUppBlock_40097814676060
// MI455X (gfx1250) — matrix operand order repaired and checked by reading, not compiled, not run on hardware
//
#include <hip/hip_runtime.h>

typedef __bf16 bf16;
typedef __attribute__((ext_vector_type(8)))  bf16  v8bf;
typedef __attribute__((ext_vector_type(16))) bf16  v16bf;
typedef __attribute__((ext_vector_type(8)))  float v8f;

#define LSEQ    2048
#define BATCH   16
#define DMODEL  1024
#define DHID    512
#define NTOK    (LSEQ * BATCH)

// ---------------------------------------------------------------------------
// CDNA5 async global->LDS (ASYNCcnt-tracked), inline asm per ISA 10.7/15.18
// vdst VGPR carries the LDS byte address; vaddr is the 64-bit global address.
// ---------------------------------------------------------------------------
__device__ __forceinline__ void async_ld_b128(const void* gptr, void* lptr) {
    unsigned lds = (unsigned)(unsigned long long)lptr;  // low 32b = LDS offset
    asm volatile("global_load_async_to_lds_b128 %0, %1, off"
                 :: "v"(lds), "v"(gptr) : "memory");
}
template <int N>
__device__ __forceinline__ void wait_asynccnt() {
    asm volatile("s_wait_asynccnt %0" :: "n"(N) : "memory");
}

// ---------------------------------------------------------------------------
// Elementwise fp32 -> bf16 convert
// ---------------------------------------------------------------------------
__global__ __launch_bounds__(256) void k_f32_to_bf16(const float* __restrict__ in,
                                                     bf16* __restrict__ out,
                                                     long long n) {
    long long i = (long long)blockIdx.x * 256 + threadIdx.x;
    long long stride = (long long)gridDim.x * 256;
    for (; i < n; i += stride) out[i] = (bf16)in[i];
}

// ---------------------------------------------------------------------------
// Transpose fp32 [R,C] -> bf16 [C,R]
// ---------------------------------------------------------------------------
__global__ __launch_bounds__(256) void k_transpose_to_bf16(const float* __restrict__ src,
                                                           bf16* __restrict__ dst,
                                                           int R, int C) {
    long long n = (long long)R * C;
    long long stride = (long long)gridDim.x * 256;
    for (long long i = (long long)blockIdx.x * 256 + threadIdx.x; i < n; i += stride) {
        int r = (int)(i / C);
        int c = (int)(i % C);
        dst[(long long)c * R + r] = (bf16)src[i];
    }
}

// ---------------------------------------------------------------------------
// Build vaT[b][p][l] = kv[(l*B+b)*1024 + 512 + p]   (bf16 -> bf16 transpose)
// ---------------------------------------------------------------------------
__global__ __launch_bounds__(256) void k_build_vaT(const bf16* __restrict__ kv,
                                                   bf16* __restrict__ vaT) {
    long long n = (long long)BATCH * DHID * LSEQ;
    long long stride = (long long)gridDim.x * 256;
    for (long long i = (long long)blockIdx.x * 256 + threadIdx.x; i < n; i += stride) {
        int l = (int)(i % LSEQ);
        long long r = i / LSEQ;
        int p = (int)(r % DHID);
        int b = (int)(r / DHID);
        vaT[i] = kv[((long long)l * BATCH + b) * (2 * DHID) + DHID + p];
    }
}

// ---------------------------------------------------------------------------
// NT GEMM: C[M,N] = alpha * A[M,K] x Bt[N,K]^T (+ addsrc)
//   A: bf16 (async double-buffered) or fp32 (sync staging w/ conversion)
//   Block: 256 threads = 8 wave32 waves, 128x128 tile; wave = 64x32 (4x2 WMMA)
// ---------------------------------------------------------------------------
template <bool A_F32, bool OUT_F32, bool ADD_SRC>
__global__ __launch_bounds__(256) void k_gemm_nt(const void* __restrict__ Av,
                                                 const bf16* __restrict__ Bt,
                                                 void* __restrict__ Cv,
                                                 const bf16* __restrict__ addsrc,
                                                 float alpha, int K,
                                                 int lda, int ldb, int ldc) {
    // padded rows: 32 bf16 data + 8 pad = 80 bytes/row (conflict-friendly)
    __shared__ bf16 As[2][128 * 40];
    __shared__ bf16 Bs[2][128 * 40];

    const int tid   = threadIdx.x;
    const int lane  = tid & 31;
    const int wave  = tid >> 5;   // 0..7
    const int waveM = wave >> 2;  // 0..1 -> 64-row half
    const int waveN = wave & 3;   // 0..3 -> 32-col quarter

    const long long rowBase = (long long)blockIdx.x * 128;
    const long long colBase = (long long)blockIdx.y * 128;

    const int sr = tid >> 1;  // staging row 0..127
    const int sh = tid & 1;   // staging half (16 elements)
    const long long aRow = (rowBase + sr) * (long long)lda;
    const long long bRow = (colBase + sr) * (long long)ldb;

    v8f acc[4][2];
#pragma unroll
    for (int mt = 0; mt < 4; ++mt)
#pragma unroll
        for (int nt = 0; nt < 2; ++nt) acc[mt][nt] = (v8f)(0.0f);

    const int fr = lane & 15;  // fragment row (m or n within 16x16 tile)
    const int fh = lane >> 4;  // fragment k-half

    // -------- per-buffer compute: fragments + 8 WMMA --------
    auto compute_tile = [&](const bf16* as, const bf16* bs) {
        v16bf aF[4], bF[2];
#pragma unroll
        for (int mt = 0; mt < 4; ++mt) {
            const bf16* p = &as[(waveM * 64 + mt * 16 + fr) * 40];
            v8bf lo = *(const v8bf*)(p + fh * 8);
            v8bf hi = *(const v8bf*)(p + 16 + fh * 8);
            aF[mt] = __builtin_shufflevector(lo, hi, 0, 1, 2, 3, 4, 5, 6, 7, 8,
                                             9, 10, 11, 12, 13, 14, 15);
        }
#pragma unroll
        for (int nt = 0; nt < 2; ++nt) {
            const bf16* p = &bs[(waveN * 32 + nt * 16 + fr) * 40];
            v8bf lo = *(const v8bf*)(p + fh * 8);
            v8bf hi = *(const v8bf*)(p + 16 + fh * 8);
            bF[nt] = __builtin_shufflevector(lo, hi, 0, 1, 2, 3, 4, 5, 6, 7, 8,
                                             9, 10, 11, 12, 13, 14, 15);
        }
#pragma unroll
        for (int mt = 0; mt < 4; ++mt)
#pragma unroll
            for (int nt = 0; nt < 2; ++nt)
                acc[mt][nt] = __builtin_amdgcn_wmma_f32_16x16x32_bf16(
                    false, aF[mt], false, bF[nt], (short)0, acc[mt][nt], false,
                    false);
    };

    if (!A_F32) {
        // ---- async double-buffered pipeline (bf16 A), branch-free steady state
        const bf16* A = (const bf16*)Av;
        auto issue = [&](int buf, int k0) {
            const bf16* ga = A + aRow + k0 + sh * 16;
            const bf16* gb = Bt + bRow + k0 + sh * 16;
            bf16* la = &As[buf][sr * 40 + sh * 16];
            bf16* lb = &Bs[buf][sr * 40 + sh * 16];
            async_ld_b128(ga, la);
            async_ld_b128(ga + 8, la + 8);
            async_ld_b128(gb, lb);
            async_ld_b128(gb + 8, lb + 8);
        };
        const int nsteps = K / 32;
        issue(0, 0);
        for (int i = 0; i < nsteps - 1; ++i) {
            issue((i + 1) & 1, (i + 1) * 32);
            wait_asynccnt<4>();  // tile i resident; tile i+1 in flight
            __syncthreads();     // all waves' tile-i loads visible
            compute_tile(As[i & 1], Bs[i & 1]);
            __syncthreads();     // done reading buf before it is re-issued
        }
        wait_asynccnt<0>();
        __syncthreads();
        compute_tile(As[(nsteps - 1) & 1], Bs[(nsteps - 1) & 1]);
    } else {
        // ---- synchronous staging with f32 -> bf16 conversion ----
        for (int k0 = 0; k0 < K; k0 += 32) {
            __syncthreads();
            {
                const float* A = (const float*)Av;
                const float4* srcA =
                    (const float4*)(A + aRow + k0 + sh * 16);
                bf16* dstA = &As[0][sr * 40 + sh * 16];
#pragma unroll
                for (int i = 0; i < 4; ++i) {
                    float4 f = srcA[i];
                    dstA[i * 4 + 0] = (bf16)f.x;
                    dstA[i * 4 + 1] = (bf16)f.y;
                    dstA[i * 4 + 2] = (bf16)f.z;
                    dstA[i * 4 + 3] = (bf16)f.w;
                }
                const uint4* srcB =
                    (const uint4*)(Bt + bRow + k0 + sh * 16);
                uint4* dstB = (uint4*)&Bs[0][sr * 40 + sh * 16];
                dstB[0] = srcB[0];
                dstB[1] = srcB[1];
            }
            __syncthreads();
            compute_tile(As[0], Bs[0]);
        }
    }

    // ---- epilogue: D 16x16 f32 layout: lane n = lane&15, m = (lane>>4)*8 + i
    const int en = lane & 15;
    const int em = (lane >> 4) * 8;
#pragma unroll
    for (int mt = 0; mt < 4; ++mt)
#pragma unroll
        for (int nt = 0; nt < 2; ++nt) {
            long long row0 = rowBase + waveM * 64 + mt * 16 + em;
            long long col  = colBase + waveN * 32 + nt * 16 + en;
#pragma unroll
            for (int i = 0; i < 8; ++i) {
                float v = acc[mt][nt][i] * alpha;
                long long off = (row0 + i) * (long long)ldc + col;
                if (ADD_SRC) v += (float)addsrc[off];
                if (OUT_F32)
                    ((float*)Cv)[off] = v;
                else
                    ((bf16*)Cv)[off] = (bf16)v;
            }
        }
}

// ---------------------------------------------------------------------------
// Row softmax, fp32, in place. One block per row.
// ---------------------------------------------------------------------------
__global__ __launch_bounds__(256) void k_softmax_rows(float* __restrict__ S, int ncols) {
    __shared__ float red[256];
    float* row = S + (long long)blockIdx.x * ncols;
    int t = threadIdx.x;

    float m = -1e30f;
    for (int c = t; c < ncols; c += 256) m = fmaxf(m, row[c]);
    red[t] = m;
    __syncthreads();
    for (int s = 128; s > 0; s >>= 1) {
        if (t < s) red[t] = fmaxf(red[t], red[t + s]);
        __syncthreads();
    }
    m = red[0];
    __syncthreads();

    float sum = 0.0f;
    for (int c = t; c < ncols; c += 256) {
        float e = __expf(row[c] - m);
        row[c] = e;
        sum += e;
    }
    red[t] = sum;
    __syncthreads();
    for (int s = 128; s > 0; s >>= 1) {
        if (t < s) red[t] += red[t + s];
        __syncthreads();
    }
    float inv = 1.0f / red[0];
    for (int c = t; c < ncols; c += 256) row[c] *= inv;
}

// ---------------------------------------------------------------------------
// SRU highway recurrence over time. One thread per (b, d) channel.
// ---------------------------------------------------------------------------
__global__ __launch_bounds__(256) void k_sru_scan(const float* __restrict__ U,
                                                  const float* __restrict__ x,
                                                  const float* __restrict__ v,
                                                  const float* __restrict__ bb,
                                                  float* __restrict__ out) {
    int idx = blockIdx.x * 256 + threadIdx.x;  // 0 .. B*DMODEL-1
    int b = idx >> 10;                         // / DMODEL
    int d = idx & (DMODEL - 1);
    float vf = v[d], vr = v[DMODEL + d];
    float bf_ = bb[d], br = bb[DMODEL + d];
    float c = 0.0f;
    for (int l = 0; l < LSEQ; ++l) {
        long long tok = (long long)l * BATCH + b;
        const float* u = U + tok * (3 * DMODEL);
        float xt = u[d];
        float fr = u[DMODEL + d];
        float rr = u[2 * DMODEL + d];
        float xres = x[tok * DMODEL + d];
        float f = 1.0f / (1.0f + __expf(-(fr + vf * c + bf_)));
        float r = 1.0f / (1.0f + __expf(-(rr + vr * c + br)));
        c = f * c + (1.0f - f) * xt;
        out[tok * DMODEL + d] = r * c + (1.0f - r) * xres;
    }
}

// ---------------------------------------------------------------------------
extern "C" void kernel_launch(void* const* d_in, const int* in_sizes, int n_in,
                              void* d_out, int out_size, void* d_ws, size_t ws_size,
                              hipStream_t stream) {
    const float* x  = (const float*)d_in[0];  // [L,B,1024]
    const float* W1 = (const float*)d_in[1];  // [1024,512]
    const float* W2 = (const float*)d_in[2];  // [512,1024]
    const float* W3 = (const float*)d_in[3];  // [512,3072]
    const float* v  = (const float*)d_in[4];  // [2,1024]
    const float* b  = (const float*)d_in[5];  // [2,1024]
    float* out = (float*)d_out;               // [L,B,1024]

    char* ws = (char*)d_ws;
    size_t off = 0;
    auto take = [&](size_t bytes) {
        char* p = ws + off;
        off += (bytes + 255) & ~(size_t)255;
        return p;
    };
    bf16*  W1T  = (bf16*)take((size_t)DHID * DMODEL * 2);          // [512,1024]
    bf16*  W2T  = (bf16*)take((size_t)(2 * DHID) * DHID * 2);      // [1024,512]
    bf16*  W3T  = (bf16*)take((size_t)(3 * DMODEL) * DHID * 2);    // [3072,512]
    bf16*  x_bf = (bf16*)take((size_t)NTOK * DMODEL * 2);          // [N,1024]
    bf16*  q_bf = (bf16*)take((size_t)NTOK * DHID * 2);            // [N,512]
    bf16*  kv_bf = (bf16*)take((size_t)NTOK * 2 * DHID * 2);       // [N,1024]
    bf16*  vaT  = (bf16*)take((size_t)BATCH * DHID * LSEQ * 2);    // [B,512,L]
    float* S    = (float*)take((size_t)LSEQ * LSEQ * 4);           // [L,L] one batch
    bf16*  cpq  = (bf16*)take((size_t)NTOK * DHID * 2);            // [N,512]
    float* U    = (float*)take((size_t)NTOK * 3 * DMODEL * 4);     // [N,3072]

    // --- precision conversion / weight transposition (bf16, N x K layouts) ---
    k_f32_to_bf16<<<2048, 256, 0, stream>>>(x, x_bf, (long long)NTOK * DMODEL);
    k_transpose_to_bf16<<<512, 256, 0, stream>>>(W1, W1T, DMODEL, DHID);
    k_transpose_to_bf16<<<512, 256, 0, stream>>>(W2, W2T, DHID, 2 * DHID);
    k_transpose_to_bf16<<<1024, 256, 0, stream>>>(W3, W3T, DHID, 3 * DMODEL);

    // --- q = x @ W1 -> bf16 [N,512] ---
    {
        dim3 g(NTOK / 128, DHID / 128);
        k_gemm_nt<false, false, false><<<g, 256, 0, stream>>>(
            x_bf, W1T, q_bf, nullptr, 1.0f, DMODEL, DMODEL, DMODEL, DHID);
    }
    // --- kv = q @ W2 -> bf16 [N,1024] ---
    {
        dim3 g(NTOK / 128, (2 * DHID) / 128);
        k_gemm_nt<false, false, false><<<g, 256, 0, stream>>>(
            q_bf, W2T, kv_bf, nullptr, 1.0f, DHID, DHID, DHID, 2 * DHID);
    }
    // --- vaT[b][p][l] ---
    k_build_vaT<<<2048, 256, 0, stream>>>(kv_bf, vaT);

    // --- attention per batch: scores -> softmax -> ctx (+q) ---
    const float scale = 0.044194173824159216f;  // 1/sqrt(512)
    for (int bb = 0; bb < BATCH; ++bb) {
        dim3 gs(LSEQ / 128, LSEQ / 128);
        k_gemm_nt<false, true, false><<<gs, 256, 0, stream>>>(
            q_bf + (size_t)bb * DHID,            // A: q_b [L,512], lda = B*512
            kv_bf + (size_t)bb * 2 * DHID,       // Bt: k_b [L,512], ldb = B*1024
            S, nullptr, scale, DHID, BATCH * DHID, BATCH * 2 * DHID, LSEQ);

        k_softmax_rows<<<LSEQ, 256, 0, stream>>>(S, LSEQ);

        dim3 gc(LSEQ / 128, DHID / 128);
        k_gemm_nt<true, false, true><<<gc, 256, 0, stream>>>(
            S,                                   // A: attn fp32 [L,L]
            vaT + (size_t)bb * DHID * LSEQ,      // Bt: vaT_b [512,L]
            cpq + (size_t)bb * DHID,             // C: (ctx+q) bf16, ldc = B*512
            q_bf + (size_t)bb * DHID,            // addsrc: q
            1.0f, LSEQ, LSEQ, LSEQ, BATCH * DHID);
    }

    // --- U = (ctx + q) @ W3 -> fp32 [N,3072] ---
    {
        dim3 g(NTOK / 128, (3 * DMODEL) / 128);
        k_gemm_nt<false, true, false><<<g, 256, 0, stream>>>(
            cpq, W3T, U, nullptr, 1.0f, DHID, DHID, DHID, 3 * DMODEL);
    }

    // --- highway recurrence ---
    k_sru_scan<<<(BATCH * DMODEL) / 256, 256, 0, stream>>>(U, x, v, b, out);
}
